// BahdanauAttention_65412351918295
// MI455X (gfx1250) — matrix operand order repaired and checked by reading, not compiled, not run on hardware
//
#include <hip/hip_runtime.h>
#include <math.h>

typedef __attribute__((ext_vector_type(16))) __bf16 v16bf;
typedef __attribute__((ext_vector_type(8)))  __bf16 v8bf;
typedef __attribute__((ext_vector_type(8)))  float  v8f;

#define NB  32      // batch
#define NS  2048    // sequence
#define DV  1024    // d_v == K of big GEMM
#define DU  1024    // units == N of big GEMM

// float -> bf16 bits, round-to-nearest-even
__device__ __forceinline__ unsigned short f2bfu(float f) {
  unsigned u = __builtin_bit_cast(unsigned, f);
  u += 0x7FFFu + ((u >> 16) & 1u);
  return (unsigned short)(u >> 16);
}
__device__ __forceinline__ __bf16 f2bf(float f) {
  return __builtin_bit_cast(__bf16, f2bfu(f));
}

// Branch-free tanh: prefer the gfx1250 hardware V_TANH_F32.
__device__ __forceinline__ float fast_tanh(float x) {
#if __has_builtin(__builtin_amdgcn_tanhf)
  return __builtin_amdgcn_tanhf(x);
#elif __has_builtin(__builtin_amdgcn_exp2f) && __has_builtin(__builtin_amdgcn_rcpf)
  // tanh(x) = 1 - 2/(exp(2x)+1); exp(2x) = exp2(x * 2*log2(e))
  float e = __builtin_amdgcn_exp2f(x * 2.885390082f);
  return 1.0f - 2.0f * __builtin_amdgcn_rcpf(e + 1.0f);
#else
  return tanhf(x);
#endif
}

// ---------------------------------------------------------------------------
// Kernel A: swizzle W1 (fp32 [K=1024][U=1024]) into bf16 WMMA-B fragment order.
// Layout: W1s[((ut*32 + kb)*32 + lane)*16 + e]
//   contiguous k order used here (A matches): K = kb*32 + (lane>>4)*16 + e, n = lane&15, u = ut*16+n.
// Each lane's 16 bf16 B-fragment elements are one contiguous 32B chunk.
// ---------------------------------------------------------------------------
__global__ __launch_bounds__(256) void w1_swizzle(const float* __restrict__ W1,
                                                  unsigned short* __restrict__ W1s) {
  int id = blockIdx.x * 256 + threadIdx.x;      // 0 .. 1048575
  int e    = id & 15;
  int lane = (id >> 4) & 31;
  int kb   = (id >> 9) & 31;
  int ut   = id >> 14;
  int K = kb * 32 + (lane >> 4) * 16 + e;
  int u = ut * 16 + (lane & 15);
  W1s[id] = f2bfu(W1[(size_t)K * DU + u]);
}

// ---------------------------------------------------------------------------
// Kernel B: pq[b][u] = query[b,:] @ W2[:,u] + b1[u] + b2[u]
// ---------------------------------------------------------------------------
__global__ __launch_bounds__(256) void pq_kernel(const float* __restrict__ query,
                                                 const float* __restrict__ W2,
                                                 const float* __restrict__ b1,
                                                 const float* __restrict__ b2,
                                                 float* __restrict__ pq) {
  __shared__ float q[DV];
  const int b = blockIdx.x >> 2;
  const int u = (blockIdx.x & 3) * 256 + threadIdx.x;
  for (int i = threadIdx.x; i < DV; i += 256) q[i] = query[(size_t)b * DV + i];
  __syncthreads();
  float acc = b1[u] + b2[u];
  #pragma unroll 8
  for (int d = 0; d < DV; ++d) acc = fmaf(q[d], W2[(size_t)d * DU + u], acc);
  pq[(size_t)b * DU + u] = acc;
}

// ---------------------------------------------------------------------------
// Kernel C (WMMA core): raw scores.
// Block = 32 s-rows of one batch, 8 waves; wave owns u-tiles w, w+8, ...
// Each B fragment (32x16 bf16, lane-contiguous) is reused by TWO WMMAs
// (s-rows 0-15 and 16-31) to halve L2 traffic on the W1 stream.
// score[b][s] = sum_u tanh( (values@W1)[s,u] + pq[b,u] ) * V[u]   (+ bV)
// ---------------------------------------------------------------------------
__global__ __launch_bounds__(256) void score_kernel(const float* __restrict__ values,
                                                    const unsigned short* __restrict__ W1s_u,
                                                    const float* __restrict__ pq,
                                                    const float* __restrict__ Vvec,
                                                    const float* __restrict__ bV,
                                                    float* __restrict__ score) {
  __shared__ __align__(16) __bf16 A_lds[32 * DV];   // 64 KB bf16 values tile
  __shared__ float red2[256 * 16];                  // 16 KB deterministic reduce
  const __bf16* W1s = reinterpret_cast<const __bf16*>(W1s_u);

  const int b   = blockIdx.x >> 6;                  // 64 s-tiles (of 32) per batch
  const int s0  = (blockIdx.x & 63) << 5;
  const int tid = threadIdx.x;

  // Stage values[b, s0:s0+32, :] (contiguous 128 KB fp32) -> bf16 LDS, row-major.
  const float4* src = (const float4*)(values + ((size_t)b * NS + s0) * DV);
  #pragma unroll 4
  for (int i = tid; i < (32 * DV) / 4; i += 256) {
    float4 v = src[i];
    int j = i << 2;
    A_lds[j + 0] = f2bf(v.x);
    A_lds[j + 1] = f2bf(v.y);
    A_lds[j + 2] = f2bf(v.z);
    A_lds[j + 3] = f2bf(v.w);
  }
  __syncthreads();

  const int wave = tid >> 5;
  const int lane = tid & 31;
  const int half = lane >> 4;       // K-group select for A/B, row-group for C/D
  const int mn   = lane & 15;       // A row (m) == C/D column (n)

  float sum0[8], sum1[8];
  #pragma unroll
  for (int i = 0; i < 8; ++i) { sum0[i] = 0.f; sum1[i] = 0.f; }

  const __bf16* Arow0 = A_lds + mn * DV + half * 16;          // rows 0..15
  const __bf16* Arow1 = A_lds + (16 + mn) * DV + half * 16;   // rows 16..31

  for (int ut = wave; ut < DU / 16; ut += 8) {
    v8f acc0 = {};
    v8f acc1 = {};
    const __bf16* Bt = W1s + (size_t)ut * (DV * 16) + lane * 16;
    #pragma unroll 2
    for (int kb = 0; kb < DV / 32; ++kb) {
      // B fragment: one contiguous, 32B-aligned lane-private chunk (shared by
      // both M-tiles below).
      v16bf bm = *(const v16bf*)(Bt + (size_t)kb * 512);
      __builtin_prefetch(Bt + (size_t)kb * 512 + 4096, 0, 1);
      // A fragments: K=k0+half*16+e
      v8bf a0lo = *(const v8bf*)(Arow0 + kb * 32);
      v8bf a0hi = *(const v8bf*)(Arow0 + kb * 32 + 8);
      v8bf a1lo = *(const v8bf*)(Arow1 + kb * 32);
      v8bf a1hi = *(const v8bf*)(Arow1 + kb * 32 + 8);
      struct P { v8bf lo, hi; };
      P p0 = {a0lo, a0hi};
      P p1 = {a1lo, a1hi};
      v16bf a0 = __builtin_bit_cast(v16bf, p0);
      v16bf a1 = __builtin_bit_cast(v16bf, p1);
      acc0 = __builtin_amdgcn_wmma_f32_16x16x32_bf16(
          false, a0, false, bm, (short)0, acc0, false, false);
      acc1 = __builtin_amdgcn_wmma_f32_16x16x32_bf16(
          false, a1, false, bm, (short)0, acc1, false, false);
    }
    // Epilogue: add pq, tanh (hardware v_tanh_f32), weight by V[u], fold over u.
    const int u = (ut << 4) + mn;
    const float pqv = pq[(size_t)b * DU + u];
    const float vn  = Vvec[u];
    #pragma unroll
    for (int v = 0; v < 8; ++v) {
      sum0[v] += fast_tanh(acc0[v] + pqv) * vn;  // C/D row = half*8 + v
      sum1[v] += fast_tanh(acc1[v] + pqv) * vn;  // C/D row = 16 + half*8 + v
    }
  }

  // Deterministic cross-lane/cross-wave reduction of the 32 row scores.
  #pragma unroll
  for (int v = 0; v < 8; ++v) {
    red2[tid * 16 + v]     = sum0[v];
    red2[tid * 16 + 8 + v] = sum1[v];
  }
  __syncthreads();
  if (tid < 32) {
    const int m    = tid;        // block-local s row 0..31
    const int tile = m >> 4;     // 0: rows 0-15 (sum0), 1: rows 16-31 (sum1)
    const int mm   = m & 15;
    float s = 0.f;
    for (int w = 0; w < 8; ++w)
      for (int n = 0; n < 16; ++n)
        s += red2[((w * 32 + (mm >> 3) * 16 + n) * 16) + tile * 8 + (mm & 7)];
    score[(size_t)b * NS + s0 + m] = s + bV[0];
  }
}

// ---------------------------------------------------------------------------
// Kernel D: in-place softmax over S (2048) per batch row.
// ---------------------------------------------------------------------------
__global__ __launch_bounds__(256) void softmax_kernel(float* __restrict__ att) {
  __shared__ float red[256];
  const int b = blockIdx.x, tid = threadIdx.x;
  float* row = att + (size_t)b * NS;
  float loc[8];
  float m = -3.402823e38f;
  #pragma unroll
  for (int i = 0; i < 8; ++i) { loc[i] = row[tid + i * 256]; m = fmaxf(m, loc[i]); }
  red[tid] = m; __syncthreads();
  for (int off = 128; off > 0; off >>= 1) {
    if (tid < off) red[tid] = fmaxf(red[tid], red[tid + off]);
    __syncthreads();
  }
  m = red[0]; __syncthreads();
  float s = 0.f;
  #pragma unroll
  for (int i = 0; i < 8; ++i) { loc[i] = __expf(loc[i] - m); s += loc[i]; }
  red[tid] = s; __syncthreads();
  for (int off = 128; off > 0; off >>= 1) {
    if (tid < off) red[tid] += red[tid + off];
    __syncthreads();
  }
  const float inv = 1.f / red[0];
  #pragma unroll
  for (int i = 0; i < 8; ++i) row[tid + i * 256] = loc[i] * inv;
}

// ---------------------------------------------------------------------------
// Kernel E: context partials. Block = (b, d-chunk of 256, s-chunk of 256).
// ---------------------------------------------------------------------------
__global__ __launch_bounds__(256) void ctx_partial(const float* __restrict__ values,
                                                   const float* __restrict__ att,
                                                   float* __restrict__ part) {
  __shared__ float wsh[256];
  const int id = blockIdx.x;
  const int sc = id & 7, dc = (id >> 3) & 3, b = id >> 5;
  const int d  = dc * 256 + threadIdx.x;
  wsh[threadIdx.x] = att[(size_t)b * NS + sc * 256 + threadIdx.x];
  __syncthreads();
  const float* vb = values + ((size_t)b * NS + sc * 256) * DV + d;
  float acc = 0.f;
  #pragma unroll 4
  for (int i = 0; i < 256; ++i) acc = fmaf(wsh[i], vb[(size_t)i * DV], acc);
  part[((size_t)b * 8 + sc) * DV + d] = acc;
}

// Kernel F: fixed-order reduction of the 8 s-chunk partials.
__global__ __launch_bounds__(256) void ctx_reduce(const float* __restrict__ part,
                                                  float* __restrict__ ctx) {
  const int idx = blockIdx.x * 256 + threadIdx.x;  // 0 .. 32767
  const int b = idx >> 10, d = idx & 1023;
  float s = 0.f;
  #pragma unroll
  for (int sc = 0; sc < 8; ++sc) s += part[((size_t)b * 8 + sc) * DV + d];
  ctx[idx] = s;
}

// ---------------------------------------------------------------------------
extern "C" void kernel_launch(void* const* d_in, const int* in_sizes, int n_in,
                              void* d_out, int out_size, void* d_ws, size_t ws_size,
                              hipStream_t stream) {
  const float* query  = (const float*)d_in[0];   // [32,1024]
  const float* values = (const float*)d_in[1];   // [32,2048,1024]
  const float* W1     = (const float*)d_in[2];   // [1024,1024]
  const float* b1     = (const float*)d_in[3];   // [1024]
  const float* W2     = (const float*)d_in[4];   // [1024,1024]
  const float* b2     = (const float*)d_in[5];   // [1024]
  const float* Vvec   = (const float*)d_in[6];   // [1024,1]
  const float* bV     = (const float*)d_in[7];   // [1]

  float* out = (float*)d_out;
  float* ctx = out;               // [32*1024] context_vector
  float* att = out + NB * DV;     // [32*2048] attention_weights (scores first)

  // Workspace: W1s (bf16, 2 MB) | pq (128 KB) | part (1 MB)  => ~3.25 MB
  char* ws = (char*)d_ws;
  unsigned short* W1s = (unsigned short*)ws;
  float* pq   = (float*)(ws + (size_t)DV * DU * 2);
  float* part = (float*)(ws + (size_t)DV * DU * 2 + (size_t)NB * DU * 4);

  hipLaunchKernelGGL(w1_swizzle,    dim3(4096), dim3(256), 0, stream, W1, W1s);
  hipLaunchKernelGGL(pq_kernel,     dim3(128),  dim3(256), 0, stream, query, W2, b1, b2, pq);
  hipLaunchKernelGGL(score_kernel,  dim3(2048), dim3(256), 0, stream, values, W1s, pq, Vvec, bV, att);
  hipLaunchKernelGGL(softmax_kernel,dim3(32),   dim3(256), 0, stream, att);
  hipLaunchKernelGGL(ctx_partial,   dim3(1024), dim3(256), 0, stream, values, att, part);
  hipLaunchKernelGGL(ctx_reduce,    dim3(128),  dim3(256), 0, stream, part, ctx);
}
